// MultiHeadAttention_25305947308246
// MI455X (gfx1250) — matrix operand order repaired and checked by reading, not compiled, not run on hardware
//
#include <hip/hip_runtime.h>

// ---------------------------------------------------------------------------
// MHA for MI455X (gfx1250): f16 WMMA (f32 accum) everywhere.
//  B=4, N=2048, C=512, H=8, D=64.  SCALE = 1/8 folded into Q.
// ---------------------------------------------------------------------------

typedef _Float16 half_t;
typedef __attribute__((ext_vector_type(16))) _Float16 v16h;
typedef __attribute__((ext_vector_type(8)))  float    v8f;

#define WMMA(a, b, c) \
  __builtin_amdgcn_wmma_f32_16x16x32_f16(false, (a), false, (b), (short)0, (c), false, false)

// A-operand (16x32 f16, M x K) from row-major memory, ld in elements.
// ISA layout: lane m=lane&15 holds row m; VGPR i pair at
//   K = (i>=4 ? 16 : 0) + (lanehi ? 8 : 0) + 2*(i&3)   (contiguous pair)
// -> lowers to two global_load_b128 per fragment.
static __device__ inline v16h load_a_frag(const half_t* __restrict__ src, int ld, int lane) {
  union { v16h v; unsigned u[8]; } r;
  const int m  = lane & 15;
  const int hi = (lane >> 4) & 1;
  const half_t* row = src + m * ld;
#pragma unroll
  for (int i = 0; i < 8; ++i) {
    const int kb = ((i & 4) << 2) + (hi << 3) + ((i & 3) << 1);
    r.u[i] = *(const unsigned*)(row + kb);
  }
  return r.v;
}

// B-operand (32x16 f16, K x N) loaded from B^T stored row-major (N x K, ld elems).
// ISA layout: lane n=lane&15 is column n; VGPR i pair at K = (i>=4?16:0) + (lanehi?8:0) + 2*(i&3).
static __device__ inline v16h load_bt_frag(const half_t* __restrict__ bt, int ld, int lane) {
  union { v16h v; unsigned u[8]; } r;
  const int n  = lane & 15;
  const int hi = (lane >> 4) & 1;
  const half_t* row = bt + n * ld;
#pragma unroll
  for (int i = 0; i < 8; ++i) {
    const int kb = ((i & 4) << 2) + (hi << 3) + ((i & 3) << 1);
    r.u[i] = *(const unsigned*)(row + kb);
  }
  return r.v;
}

// ---------------------------------------------------------------------------
__global__ __launch_bounds__(256) void cast_f32_to_f16(const float* __restrict__ s,
                                                       half_t* __restrict__ d, int n) {
  int i = blockIdx.x * 256 + threadIdx.x;
  if (i < n) d[i] = (half_t)s[i];
}

// ---------------------------------------------------------------------------
// QKV GEMM: Y[8192][1536] = X[8192][512] @ W^T + b, W row-major (1536x512) == B^T.
// Wave tile 32x64, 8 waves/WG as 4(M)x2(N) -> 128x128.  grid (64, 12).
// Epilogue scatters: cols [0,512)->Q (x SCALE), [512,1024)->K, [1024,1536)->V^T.
__global__ __launch_bounds__(256) void qkv_gemm(const half_t* __restrict__ X,
                                                const half_t* __restrict__ W,
                                                const float*  __restrict__ bias,
                                                half_t* __restrict__ Q,
                                                half_t* __restrict__ K,
                                                half_t* __restrict__ VT) {
  const int lane = threadIdx.x & 31;
  const int wv   = threadIdx.x >> 5;
  const int m0   = blockIdx.x * 128 + (wv >> 1) * 32;
  const int n0   = blockIdx.y * 128 + (wv & 1) * 64;

  v8f zero = {};
  v8f acc[2][4];
#pragma unroll
  for (int i = 0; i < 2; ++i)
#pragma unroll
    for (int j = 0; j < 4; ++j) acc[i][j] = zero;

  for (int k = 0; k < 512; k += 32) {
    v16h a0 = load_a_frag(X + m0 * 512 + k, 512, lane);
    v16h a1 = load_a_frag(X + (m0 + 16) * 512 + k, 512, lane);
#pragma unroll
    for (int nt = 0; nt < 4; ++nt) {
      v16h b = load_bt_frag(W + (n0 + nt * 16) * 512 + k, 512, lane);
      acc[0][nt] = WMMA(a0, b, acc[0][nt]);
      acc[1][nt] = WMMA(a1, b, acc[1][nt]);
    }
  }

  const int rb = (lane >> 4) * 8;   // C/D layout: half-wave row base
  const int nl = lane & 15;
#pragma unroll
  for (int mt = 0; mt < 2; ++mt)
#pragma unroll
    for (int nt = 0; nt < 4; ++nt) {
      const int   col = n0 + nt * 16 + nl;
      const float bv  = bias[col];
      const int   seg = col >> 9;       // wave-uniform (tiles don't straddle 512)
      const int   j   = col & 511;
      const int   h   = j >> 6, dd = j & 63;
#pragma unroll
      for (int r = 0; r < 8; ++r) {
        const int row = m0 + mt * 16 + rb + r;
        const int b_  = row >> 11, n_ = row & 2047;
        const float v = acc[mt][nt][r] + bv;
        const int  bh = b_ * 8 + h;
        if (seg == 0)      Q [(bh * 2048 + n_) * 64 + dd] = (half_t)(v * 0.125f);
        else if (seg == 1) K [(bh * 2048 + n_) * 64 + dd] = (half_t)v;
        else               VT[(bh * 64 + dd) * 2048 + n_] = (half_t)v;
      }
    }
}

// ---------------------------------------------------------------------------
// Flash attention: one wave per (b,h, 16-row q tile).  4 waves / block.
// 64-key blocks: 16 WMMAs per block against a fixed 64-shuffle softmax cost,
// halving the per-key VALU overhead vs 32-key blocks.
// Q pre-scaled by 1/sqrt(D).  K row-major [n][d], V^T [d][n].
__global__ __launch_bounds__(128) void attn_kernel(const half_t* __restrict__ Q,
                                                   const half_t* __restrict__ K,
                                                   const half_t* __restrict__ VT,
                                                   half_t* __restrict__ O) {
  __shared__ half_t plds[4][16 * 64];
  const int lane = threadIdx.x & 31;
  const int wv   = threadIdx.x >> 5;
  const int gw   = blockIdx.x * 4 + wv;       // 0..4095
  const int qt   = gw & 127;                  // q tile (16 rows)
  const int bh   = gw >> 7;                   // 0..31
  const int b_   = bh >> 3, h_ = bh & 7;

  const half_t* qp = Q  + (bh * 2048 + qt * 16) * 64;
  const half_t* kp = K  + bh * 2048 * 64;
  const half_t* vp = VT + bh * 64 * 2048;

  const v16h qf0 = load_a_frag(qp,      64, lane);   // d 0..31
  const v16h qf1 = load_a_frag(qp + 32, 64, lane);   // d 32..63

  v8f zero = {};
  float mrow[8], lrow[8];
  v8f oacc[4];
#pragma unroll
  for (int r = 0; r < 8; ++r) { mrow[r] = -3.0e38f; lrow[r] = 0.f; }
#pragma unroll
  for (int dt = 0; dt < 4; ++dt) oacc[dt] = zero;

  half_t* pw = plds[wv];
  const int nl = lane & 15;
  const int rb = (lane >> 4) * 8;

  for (int kb = 0; kb < 2048; kb += 64) {
    // ---- S = Q K^T for 64 keys (four 16-col subtiles) ----
    v8f s[4];
#pragma unroll
    for (int kk = 0; kk < 4; ++kk) {
      const half_t* kr = kp + (kb + kk * 16) * 64;
      v16h b0 = load_bt_frag(kr,      64, lane);
      v16h b1 = load_bt_frag(kr + 32, 64, lane);
      v8f z = zero;
      z = WMMA(qf0, b0, z);
      z = WMMA(qf1, b1, z);
      s[kk] = z;
    }
    // ---- online softmax: per-row max/sum across the 16-lane half ----
    float pscale[8];
#pragma unroll
    for (int r = 0; r < 8; ++r) {
      float tm = fmaxf(fmaxf(s[0][r], s[1][r]), fmaxf(s[2][r], s[3][r]));
#pragma unroll
      for (int off = 1; off < 16; off <<= 1) tm = fmaxf(tm, __shfl_xor(tm, off, 32));
      const float mn = fmaxf(mrow[r], tm);
      pscale[r] = __expf(mrow[r] - mn);
      mrow[r]   = mn;
      float ps = 0.f;
#pragma unroll
      for (int kk = 0; kk < 4; ++kk) {
        const float e = __expf(s[kk][r] - mn);
        s[kk][r] = e;
        ps += e;
      }
#pragma unroll
      for (int off = 1; off < 16; off <<= 1) ps += __shfl_xor(ps, off, 32);
      lrow[r] = lrow[r] * pscale[r] + ps;
    }
    // ---- P: C/D layout -> A layout via LDS (16x64 f16) ----
#pragma unroll
    for (int kk = 0; kk < 4; ++kk)
#pragma unroll
      for (int r = 0; r < 8; ++r)
        pw[(rb + r) * 64 + kk * 16 + nl] = (half_t)s[kk][r];
    asm volatile("s_wait_dscnt 0x0" ::: "memory");  // LDS store->load ordering
    const v16h pf0 = load_a_frag(pw,      64, lane);  // keys kb..kb+31
    const v16h pf1 = load_a_frag(pw + 32, 64, lane);  // keys kb+32..kb+63
    // ---- O = O*pscale + P V ----
#pragma unroll
    for (int dt = 0; dt < 4; ++dt) {
#pragma unroll
      for (int r = 0; r < 8; ++r) oacc[dt][r] *= pscale[r];
      v16h vf0 = load_bt_frag(vp + (dt * 16) * 2048 + kb,      2048, lane);
      v16h vf1 = load_bt_frag(vp + (dt * 16) * 2048 + kb + 32, 2048, lane);
      oacc[dt] = WMMA(pf0, vf0, oacc[dt]);
      oacc[dt] = WMMA(pf1, vf1, oacc[dt]);
    }
  }

#pragma unroll
  for (int r = 0; r < 8; ++r) lrow[r] = 1.0f / lrow[r];
#pragma unroll
  for (int dt = 0; dt < 4; ++dt)
#pragma unroll
    for (int r = 0; r < 8; ++r) {
      const int n_ = qt * 16 + rb + r;
      const int d_ = dt * 16 + nl;
      O[(b_ * 2048 + n_) * 512 + h_ * 64 + d_] = (half_t)(oacc[dt][r] * lrow[r]);
    }
}

// ---------------------------------------------------------------------------
// Output projection: out[8192][512] = Oh @ proj_w^T + b  (f32 stores)
__global__ __launch_bounds__(256) void proj_gemm(const half_t* __restrict__ A,
                                                 const half_t* __restrict__ W,
                                                 const float*  __restrict__ bias,
                                                 float* __restrict__ out) {
  const int lane = threadIdx.x & 31;
  const int wv   = threadIdx.x >> 5;
  const int m0   = blockIdx.x * 128 + (wv >> 1) * 32;
  const int n0   = blockIdx.y * 128 + (wv & 1) * 64;

  v8f zero = {};
  v8f acc[2][4];
#pragma unroll
  for (int i = 0; i < 2; ++i)
#pragma unroll
    for (int j = 0; j < 4; ++j) acc[i][j] = zero;

  for (int k = 0; k < 512; k += 32) {
    v16h a0 = load_a_frag(A + m0 * 512 + k, 512, lane);
    v16h a1 = load_a_frag(A + (m0 + 16) * 512 + k, 512, lane);
#pragma unroll
    for (int nt = 0; nt < 4; ++nt) {
      v16h b = load_bt_frag(W + (n0 + nt * 16) * 512 + k, 512, lane);
      acc[0][nt] = WMMA(a0, b, acc[0][nt]);
      acc[1][nt] = WMMA(a1, b, acc[1][nt]);
    }
  }

  const int rb = (lane >> 4) * 8;
  const int nl = lane & 15;
#pragma unroll
  for (int mt = 0; mt < 2; ++mt)
#pragma unroll
    for (int nt = 0; nt < 4; ++nt) {
      const int   col = n0 + nt * 16 + nl;
      const float bv  = bias[col];
#pragma unroll
      for (int r = 0; r < 8; ++r) {
        const int row = m0 + mt * 16 + rb + r;
        out[row * 512 + col] = acc[mt][nt][r] + bv;
      }
    }
}

// ---------------------------------------------------------------------------
extern "C" void kernel_launch(void* const* d_in, const int* in_sizes, int n_in,
                              void* d_out, int out_size, void* d_ws, size_t ws_size,
                              hipStream_t stream) {
  (void)in_sizes; (void)n_in; (void)out_size; (void)ws_size;
  const float* x      = (const float*)d_in[0];
  const float* qkv_w  = (const float*)d_in[1];
  const float* qkv_b  = (const float*)d_in[2];
  const float* proj_w = (const float*)d_in[3];
  const float* proj_b = (const float*)d_in[4];
  float* out = (float*)d_out;

  // workspace layout (bytes), total ~44.0 MB
  char* ws = (char*)d_ws;
  half_t* xh  = (half_t*)ws;  ws += (size_t)8192 * 512 * 2;
  half_t* wqh = (half_t*)ws;  ws += (size_t)1536 * 512 * 2;
  half_t* wph = (half_t*)ws;  ws += (size_t)512 * 512 * 2;
  half_t* qws = (half_t*)ws;  ws += (size_t)8192 * 512 * 2;
  half_t* kws = (half_t*)ws;  ws += (size_t)8192 * 512 * 2;
  half_t* vtw = (half_t*)ws;  ws += (size_t)8192 * 512 * 2;
  half_t* ows = (half_t*)ws;  ws += (size_t)8192 * 512 * 2;

  cast_f32_to_f16<<<16384, 256, 0, stream>>>(x, xh, 8192 * 512);
  cast_f32_to_f16<<<3072,  256, 0, stream>>>(qkv_w, wqh, 1536 * 512);
  cast_f32_to_f16<<<1024,  256, 0, stream>>>(proj_w, wph, 512 * 512);

  qkv_gemm<<<dim3(64, 12), 256, 0, stream>>>(xh, wqh, qkv_b, qws, kws, vtw);
  attn_kernel<<<1024, 128, 0, stream>>>(qws, kws, vtw, ows);
  proj_gemm<<<dim3(64, 4), 256, 0, stream>>>(ows, wph, proj_b, out);
}
